// CrossAttention_45457933861287
// MI455X (gfx1250) — matrix operand order repaired and checked by reading, not compiled, not run on hardware
//
#include <hip/hip_runtime.h>

#define NN 8
#define SS 1024
#define EMBD 512
#define HH 8
#define DD 64

typedef __attribute__((ext_vector_type(16))) __bf16 v16bf;
typedef __attribute__((ext_vector_type(8)))  __bf16 v8bf;
typedef __attribute__((ext_vector_type(8)))  float  v8f;

// ---- Tensor Data Mover availability (probe-guarded; sync fallback otherwise) ----
#if defined(__HIP_DEVICE_COMPILE__) && __has_builtin(__builtin_amdgcn_tensor_load_to_lds) && \
    __has_builtin(__builtin_amdgcn_s_wait_tensorcnt)
#define HAVE_TDM 1
#else
#define HAVE_TDM 0
#endif

#if HAVE_TDM
typedef __attribute__((ext_vector_type(4))) unsigned tdm_u4;
typedef __attribute__((ext_vector_type(8))) int      tdm_i8;
typedef __attribute__((ext_vector_type(4))) int      tdm_i4;

// Issue one TDM 2D tile load (ISA 8.3/8.4 D# layout): tile_d0 x tile_d1 elements of
// 2 bytes, row stride stride_d0 elements, packed contiguously into LDS at lds_off.
__device__ __forceinline__ void tdm_load_tile_2d(const void* gaddr, unsigned lds_off,
                                                 int tile_d0, int tile_d1,
                                                 int tensor_d0, int tensor_d1,
                                                 int stride_d0) {
  unsigned long long ga = (unsigned long long)(size_t)gaddr;
  tdm_u4 g0;
  g0[0] = 1u;                                                  // count=1 (valid user D#)
  g0[1] = lds_off;                                             // lds_addr (bytes)
  g0[2] = (unsigned)(ga & 0xFFFFFFFFu);                        // global_addr[31:0]
  g0[3] = (unsigned)((ga >> 32) & 0x1FFFFFFu) | 0x80000000u;   // global_addr[56:32] | type=2
  tdm_i8 g1;
  g1[0] = 1 << 16;                                             // data_size=1 (2B), mask=0
  g1[1] = (tensor_d0 & 0xFFFF) << 16;                          // tensor_dim0 low16 @bit48
  g1[2] = ((tensor_d0 >> 16) & 0xFFFF) | ((tensor_d1 & 0xFFFF) << 16);
  g1[3] = ((tensor_d1 >> 16) & 0xFFFF) | (tile_d0 << 16);      // tile_dim0 @[127:112]
  g1[4] = tile_d1 & 0xFFFF;                                    // tile_dim1 (tile_dim2=0)
  g1[5] = stride_d0;                                           // tensor_dim0_stride low32
  g1[6] = 0;
  g1[7] = 0;                                                   // tensor_dim1_stride (2D)
  tdm_i4 z4 = {0, 0, 0, 0};
#if __clang_major__ >= 23
  tdm_i8 z8 = {0, 0, 0, 0, 0, 0, 0, 0};
  __builtin_amdgcn_tensor_load_to_lds(g0, g1, z4, z4, z8, 0);
#else
  __builtin_amdgcn_tensor_load_to_lds(g0, g1, z4, z4, 0);
#endif
}
#endif

// ---------- scalar bf16 helpers (round-to-nearest-even) ----------
__device__ __forceinline__ __bf16 f2bf(float f) {
  unsigned u = __builtin_bit_cast(unsigned, f);
  unsigned short s = (unsigned short)((u + 0x7FFFu + ((u >> 16) & 1u)) >> 16);
  return __builtin_bit_cast(__bf16, s);
}

__device__ __forceinline__ v8f wmma_bf16(v16bf a, v16bf b, v8f c) {
  // D = A(16x32) * B(32x16) + C, f32 accumulate
  return __builtin_amdgcn_wmma_f32_16x16x32_bf16(false, a, false, b, (short)0, c, false, false);
}

// ---------- fragment loaders: contiguous k order K = (L>>4)*16 + i, shared by A and B ----------
// A-operand (16x32 bf16): lane L holds row M=L&15; K = (L>>4)*16 + i
// -> two contiguous 8-element (16B) chunks at col offsets {(L>>4)*16, (L>>4)*16+8}.
__device__ __forceinline__ v16bf load_a_frag(const __bf16* base, int stride, int lane, int kb) {
  const __bf16* r = base + (size_t)(lane & 15) * stride + kb + ((lane >> 4) << 4);
  v8bf lo = *(const v8bf*)(r);
  v8bf hi = *(const v8bf*)(r + 8);
  return __builtin_shufflevector(lo, hi, 0,1,2,3,4,5,6,7,8,9,10,11,12,13,14,15);
}

__device__ __forceinline__ v16bf load_a_frag_f32(const float* base, int stride, int lane, int kb) {
  const float* r = base + (size_t)(lane & 15) * stride + kb + ((lane >> 4) << 4);
  v16bf o;
#pragma unroll
  for (int i = 0; i < 8; ++i) o[i] = f2bf(r[i]);
#pragma unroll
  for (int i = 0; i < 8; ++i) o[8 + i] = f2bf(r[8 + i]);
  return o;
}

// B-operand (32x16 bf16), element (K,N) stored at tile[N*stride + K]:
// lane L holds column N=L&15; K = kb + (L>>4)*16 + i, 16 contiguous elements (32B).
__device__ __forceinline__ v16bf load_bT_frag(const __bf16* tile, int stride, int lane, int kb) {
  const __bf16* r = tile + (size_t)(lane & 15) * stride + kb + ((lane >> 4) << 4);
  return *(const v16bf*)r;
}

// B-operand from the relative-position table E (row-major [S][64]), with row clamp
// (clamped rows only feed masked (s>q) entries of the skewed bias tile).
__device__ __forceinline__ v16bf load_e_frag(const __bf16* Eb, int jbase, int lane, int kb) {
  int j = jbase + (lane & 15);
  j = j < 0 ? 0 : (j > SS - 1 ? SS - 1 : j);
  const __bf16* r = Eb + (size_t)j * DD + kb + ((lane >> 4) << 4);
  return *(const v16bf*)r;
}

// ---------- stage 0a: f32 -> bf16 weight/table conversion ----------
__global__ void cvt_bf16(const float* __restrict__ src, __bf16* __restrict__ dst, int n) {
  int i = blockIdx.x * blockDim.x + threadIdx.x;
  if (i < n) dst[i] = f2bf(src[i]);
}

// ---------- stage 0b: per-head projections  out = X(16x64) * W^T + b ----------
__global__ __launch_bounds__(32) void proj_kernel(const float* __restrict__ X,
                                                  const __bf16* __restrict__ W,
                                                  const float* __restrict__ bias,
                                                  __bf16* __restrict__ out) {
  int lane = threadIdx.x;
  int r0 = blockIdx.x * 16;
  const float* abase = X + (size_t)r0 * DD;
  v16bf a0 = load_a_frag_f32(abase, DD, lane, 0);
  v16bf a1 = load_a_frag_f32(abase, DD, lane, 32);
#pragma unroll
  for (int dc = 0; dc < 4; ++dc) {
    v8f c = {};
    v16bf b0 = load_bT_frag(W + dc * 16 * DD, DD, lane, 0);
    v16bf b1 = load_bT_frag(W + dc * 16 * DD, DD, lane, 32);
    c = wmma_bf16(a0, b0, c);
    c = wmma_bf16(a1, b1, c);
    int col = dc * 16 + (lane & 15);
    float bb = bias[col];
#pragma unroll
    for (int vv = 0; vv < 8; ++vv) {
      int row = r0 + ((lane >> 4) << 3) + vv;   // C layout: M = (L>>4)*8 + v
      out[(size_t)row * DD + col] = f2bf(c[vv] + bb);
    }
  }
}

// ---------- stage 1: fused flash attention + skew bias ----------
// grid = N*H*(S/64), block = 128 (4 waves). Wave w owns q rows [q0, q0+16).
// z[q] = (1/l[q]) * sum_{s>q} exp(qk*scale - m) * vals[s] + sum_{s<=q} (vals[q].E[S-1-q+s]) * vals[s]
__global__ __launch_bounds__(128) void attn_kernel(const __bf16* __restrict__ Q,
                                                   const __bf16* __restrict__ K,
                                                   const __bf16* __restrict__ V,
                                                   const __bf16* __restrict__ Eb,
                                                   __bf16* __restrict__ Zout) {
  __shared__ __bf16 Klds[32 * 64];     // K tile, row-major [s'][d] (TDM destination)
  __shared__ __bf16 Vt[64 * 32];       // V tile transposed [d][s'] (B-operand layout)
  __shared__ __bf16 Plds[4][16 * 32];  // per-wave P / bias tile (C-layout -> A-layout bounce)
  __shared__ __bf16 QElds[4][16 * 48]; // per-wave QE window for the skew remap

  int lane = threadIdx.x & 31;
  int w = threadIdx.x >> 5;
  int b = blockIdx.x;
  int qblk = b & 15;         // S/64 = 16 q-blocks
  int h = (b >> 4) & 7;
  int n = b >> 7;
  int q0 = qblk * 64 + w * 16;

  const __bf16* Qbase = Q + ((size_t)n * SS) * EMBD + h * DD;  // row stride EMBD
  const __bf16* Kbase = K + ((size_t)n * SS) * EMBD + h * DD;
  const __bf16* Vbase = V + ((size_t)n * SS) * EMBD + h * DD;

#if HAVE_TDM
  unsigned kldsOff = (unsigned)(size_t)(__attribute__((address_space(3))) __bf16*)Klds;
#endif

  v16bf aq0 = load_a_frag(Qbase + (size_t)q0 * EMBD, EMBD, lane, 0);
  v16bf aq1 = load_a_frag(Qbase + (size_t)q0 * EMBD, EMBD, lane, 32);
  v16bf av0 = load_a_frag(Vbase + (size_t)q0 * EMBD, EMBD, lane, 0);
  v16bf av1 = load_a_frag(Vbase + (size_t)q0 * EMBD, EMBD, lane, 32);

  float m[8], l[8];
#pragma unroll
  for (int vv = 0; vv < 8; ++vv) { m[vv] = -1e30f; l[vv] = 0.0f; }
  v8f accS[4] = {{}, {}, {}, {}};
  v8f accB[4] = {{}, {}, {}, {}};

  const float scale = 0.044194173824159216f;  // 1/sqrt(EMB)

  for (int s0 = 0; s0 < SS; s0 += 32) {
    __syncthreads();  // previous iteration's readers done with Klds/Vt
    {
      int t = threadIdx.x;
      int sr = t >> 2;              // 0..31: s' within tile
      int cb = (t & 3) * 16;        // 0,16,32,48: d chunk
      const __bf16* vrow = Vbase + (size_t)(s0 + sr) * EMBD + cb;
      v16bf vvec = *(const v16bf*)vrow;
#pragma unroll
      for (int j = 0; j < 16; ++j) Vt[(cb + j) * 32 + sr] = vvec[j];  // transpose into LDS
      if (s0 + 32 < SS) __builtin_prefetch(vrow + 32 * EMBD, 0, 1);
#if HAVE_TDM
      // K tile via Tensor Data Mover: one 2D descriptor issued by wave 0, tracked
      // with TENSORcnt; rows of 64 bf16 packed at stride 128B to match Klds.
      if (w == 0) {
        tdm_load_tile_2d(Kbase + (size_t)s0 * EMBD, kldsOff,
                         /*tile*/ DD, 32, /*tensor*/ DD, SS - s0, /*stride*/ EMBD);
        __builtin_amdgcn_s_wait_tensorcnt(0);
      }
#else
      const __bf16* krow = Kbase + (size_t)(s0 + sr) * EMBD + cb;
      *(v16bf*)&Klds[sr * 64 + cb] = *(const v16bf*)krow;
      if (s0 + 32 < SS) __builtin_prefetch(krow + 32 * EMBD, 0, 1);
#endif
    }
    __syncthreads();

    // last q-tile runs softmax over all s so the fully-masked row q=S-1 gets the
    // reference's uniform 1/S distribution; otherwise only tiles containing s>q.
    bool doSoft = (s0 + 31 > q0) || (q0 == SS - 16);
    bool doBias = (s0 <= q0 + 15);

    if (doSoft) {
      v8f c0 = {}, c1 = {};
      c0 = wmma_bf16(aq0, load_bT_frag(Klds, 64, lane, 0), c0);
      c0 = wmma_bf16(aq1, load_bT_frag(Klds, 64, lane, 32), c0);
      c1 = wmma_bf16(aq0, load_bT_frag(Klds + 16 * 64, 64, lane, 0), c1);
      c1 = wmma_bf16(aq1, load_bT_frag(Klds + 16 * 64, 64, lane, 32), c1);

      int qgb = q0 + ((lane >> 4) << 3);
      int sg0 = s0 + (lane & 15);
      int sg1 = sg0 + 16;
#pragma unroll
      for (int vv = 0; vv < 8; ++vv) {
        int qg = qgb + vv;
        float x0 = (sg0 > qg) ? c0[vv] * scale : -1e30f;  // keep strictly-upper triangle
        float x1 = (sg1 > qg) ? c1[vv] * scale : -1e30f;
        float mt = fmaxf(x0, x1);
#pragma unroll
        for (int off = 1; off < 16; off <<= 1) mt = fmaxf(mt, __shfl_xor(mt, off, 32));
        float mn = fmaxf(m[vv], mt);
        float corr = __expf(m[vv] - mn);
        float p0 = __expf(x0 - mn);
        float p1 = __expf(x1 - mn);
        float rs = p0 + p1;
#pragma unroll
        for (int off = 1; off < 16; off <<= 1) rs += __shfl_xor(rs, off, 32);
        l[vv] = l[vv] * corr + rs;
        m[vv] = mn;
#pragma unroll
        for (int f = 0; f < 4; ++f) accS[f][vv] *= corr;
        int M = ((lane >> 4) << 3) + vv;
        Plds[w][M * 32 + (lane & 15)] = f2bf(p0);        // C-layout -> LDS
        Plds[w][M * 32 + 16 + (lane & 15)] = f2bf(p1);
      }
      v16bf ap = load_a_frag(&Plds[w][0], 32, lane, 0);  // re-read as A-operand
#pragma unroll
      for (int dc = 0; dc < 4; ++dc)
        accS[dc] = wmma_bf16(ap, load_bT_frag(Vt + dc * 16 * 32, 32, lane, 0), accS[dc]);
    }

    if (doBias) {
      // QE window: j = S-1-(q-s) = jb + (15 + si - qi), width 47 -> compute 16x48 via WMMA.
      int jb = SS - 16 - q0 + s0;
#pragma unroll
      for (int jc = 0; jc < 3; ++jc) {
        v8f c = {};
        c = wmma_bf16(av0, load_e_frag(Eb, jb + jc * 16, lane, 0), c);
        c = wmma_bf16(av1, load_e_frag(Eb, jb + jc * 16, lane, 32), c);
#pragma unroll
        for (int vv = 0; vv < 8; ++vv) {
          int M = ((lane >> 4) << 3) + vv;
          QElds[w][M * 48 + jc * 16 + (lane & 15)] = f2bf(c[vv]);
        }
      }
      // skew remap: Bias[qi][si] = QE[qi][15+si-qi] when s<=q, else 0
      for (int idx = lane; idx < 16 * 32; idx += 32) {
        int qi = idx >> 5, si = idx & 31;
        bool keep = (s0 + si) <= (q0 + qi);
        Plds[w][qi * 32 + si] = keep ? QElds[w][qi * 48 + 15 + si - qi] : f2bf(0.0f);
      }
      v16bf ap2 = load_a_frag(&Plds[w][0], 32, lane, 0);
#pragma unroll
      for (int dc = 0; dc < 4; ++dc)
        accB[dc] = wmma_bf16(ap2, load_bT_frag(Vt + dc * 16 * 32, 32, lane, 0), accB[dc]);
    }
  }

  // epilogue: z = accS / l + accB   (bias part is NOT softmax-normalized)
  __bf16* zrow = Zout + ((size_t)n * SS) * EMBD + h * DD;
#pragma unroll
  for (int vv = 0; vv < 8; ++vv) {
    float rl = 1.0f / l[vv];
    int row = q0 + ((lane >> 4) << 3) + vv;
#pragma unroll
    for (int dc = 0; dc < 4; ++dc) {
      int col = dc * 16 + (lane & 15);
      zrow[(size_t)row * EMBD + col] = f2bf(accS[dc][vv] * rl + accB[dc][vv]);
    }
  }
}

// ---------- stage 2: output projection  out = Z(8192x512) * Wo^T + bo ----------
__global__ __launch_bounds__(256) void outproj_kernel(const __bf16* __restrict__ Z,
                                                      const __bf16* __restrict__ Wo,
                                                      const float* __restrict__ bo,
                                                      float* __restrict__ out) {
  int lane = threadIdx.x & 31;
  int w = threadIdx.x >> 5;      // 8 waves, 4 output 16-col chunks each
  int r0 = blockIdx.x * 16;
  const __bf16* abase = Z + (size_t)r0 * EMBD;
  v8f c[4] = {{}, {}, {}, {}};
  for (int kf = 0; kf < 16; ++kf) {
    v16bf a = load_a_frag(abase, EMBD, lane, kf * 32);
#pragma unroll
    for (int i = 0; i < 4; ++i) {
      int ec = w * 4 + i;
      v16bf bfr = load_bT_frag(Wo + (size_t)ec * 16 * EMBD, EMBD, lane, kf * 32);
      c[i] = wmma_bf16(a, bfr, c[i]);
    }
  }
#pragma unroll
  for (int i = 0; i < 4; ++i) {
    int col = (w * 4 + i) * 16 + (lane & 15);
    float bb = bo[col];
#pragma unroll
    for (int vv = 0; vv < 8; ++vv) {
      int row = r0 + ((lane >> 4) << 3) + vv;
      out[(size_t)row * EMBD + col] = c[i][vv] + bb;
    }
  }
}

// ---------- host launcher ----------
extern "C" void kernel_launch(void* const* d_in, const int* in_sizes, int n_in,
                              void* d_out, int out_size, void* d_ws, size_t ws_size,
                              hipStream_t stream) {
  const float* v  = (const float*)d_in[0];
  const float* k  = (const float*)d_in[1];
  const float* q  = (const float*)d_in[2];
  const float* Wv = (const float*)d_in[3];
  const float* bv = (const float*)d_in[4];
  const float* Wk = (const float*)d_in[5];
  const float* bk = (const float*)d_in[6];
  const float* Wq = (const float*)d_in[7];
  const float* bq = (const float*)d_in[8];
  const float* E  = (const float*)d_in[9];
  const float* Wo = (const float*)d_in[10];
  const float* bo = (const float*)d_in[11];
  float* out = (float*)d_out;

  char* ws = (char*)d_ws;
  size_t off = 0;
  auto carve = [&](size_t bytes) -> void* {
    void* p = ws + off;
    off += (bytes + 255) & ~(size_t)255;
    return p;
  };
  const size_t tok = (size_t)NN * SS * EMBD;  // 4,194,304 elements
  __bf16* Wv_b   = (__bf16*)carve((size_t)DD * DD * 2);
  __bf16* Wk_b   = (__bf16*)carve((size_t)DD * DD * 2);
  __bf16* Wq_b   = (__bf16*)carve((size_t)DD * DD * 2);
  __bf16* Wo_b   = (__bf16*)carve((size_t)EMBD * EMBD * 2);
  __bf16* E_b    = (__bf16*)carve((size_t)SS * DD * 2);
  __bf16* vals_b = (__bf16*)carve(tok * 2);
  __bf16* keys_b = (__bf16*)carve(tok * 2);
  __bf16* qrs_b  = (__bf16*)carve(tok * 2);
  __bf16* z_b    = (__bf16*)carve(tok * 2);

  cvt_bf16<<<dim3(16),   256, 0, stream>>>(Wv, Wv_b, DD * DD);
  cvt_bf16<<<dim3(16),   256, 0, stream>>>(Wk, Wk_b, DD * DD);
  cvt_bf16<<<dim3(16),   256, 0, stream>>>(Wq, Wq_b, DD * DD);
  cvt_bf16<<<dim3(1024), 256, 0, stream>>>(Wo, Wo_b, EMBD * EMBD);
  cvt_bf16<<<dim3(256),  256, 0, stream>>>(E, E_b, SS * DD);

  const int R = NN * SS * HH;  // 65536 projection rows
  proj_kernel<<<dim3(R / 16), 32, 0, stream>>>(v, Wv_b, bv, vals_b);
  proj_kernel<<<dim3(R / 16), 32, 0, stream>>>(k, Wk_b, bk, keys_b);
  proj_kernel<<<dim3(R / 16), 32, 0, stream>>>(q, Wq_b, bq, qrs_b);

  attn_kernel<<<dim3(NN * HH * (SS / 64)), 128, 0, stream>>>(qrs_b, keys_b, vals_b, E_b, z_b);

  outproj_kernel<<<dim3(NN * SS / 16), 256, 0, stream>>>(z_b, Wo_b, bo, out);
}
